// TAdaConv2d_38156489458015
// MI455X (gfx1250) — matrix operand order repaired and checked by reading, not compiled, not run on hardware
//
#include <hip/hip_runtime.h>

// Problem constants (match reference)
#define B_    8
#define CIN_  256
#define COUT_ 256
#define T_    16
#define H_    28
#define W_    28
#define HW_   (H_ * W_)          // 784
#define PIXTILE 64
#define NTILES 13                // ceil(784/64)
#define K9    9                  // 3x3 taps
#define CBLK  32                 // K step (f16 WMMA K=32)
#define NCB   (CIN_ / CBLK)      // 8 ci-blocks
#define NSTEP (K9 * NCB)         // 72 k-steps
#define LDSPAD 40                // padded row length (halfs), 80B = 16B aligned

typedef __attribute__((ext_vector_type(16))) _Float16 v16h;
typedef __attribute__((ext_vector_type(8)))  _Float16 v8h;
typedef __attribute__((ext_vector_type(8)))  float    v8f;
typedef __attribute__((ext_vector_type(4)))  unsigned int u32x4;
typedef __attribute__((ext_vector_type(8)))  unsigned int u32x8;

// ---------------------------------------------------------------------------
// One-shot weight repack: f32 OIHW -> f16 tiles [tap*8+cb][co][kk] (16KB each)
// out index: ((k9*8 + cb)*COUT + co)*32 + kk   with ci = cb*32 + kk
// ---------------------------------------------------------------------------
__global__ __launch_bounds__(256) void repack_weights_f16(
    const float* __restrict__ w, _Float16* __restrict__ wp) {
  const int idx = blockIdx.x * 256 + threadIdx.x;   // 0 .. 9*8*256*32-1
  const int kk  = idx & 31;
  const int co  = (idx >> 5) & 255;
  const int blk = idx >> 13;                        // 0..71
  const int cb  = blk & 7;
  const int k9  = blk >> 3;
  const int ci  = cb * 32 + kk;
  const int kh  = k9 / 3;
  const int kw  = k9 % 3;
  const float v = w[((co * CIN_ + ci) * 3 + kh) * 3 + kw];
  wp[idx] = (_Float16)v;
}

// ---------------------------------------------------------------------------
// TDM: DMA one 16KB weight tile (4096 dwords) global -> LDS, inserting a
// 16B pad after every 64B so LDS lands in the bank-padded [256][40]-half
// layout the WMMA fragment reads use.
// D# per CDNA5 ISA ch.8:
//   group0: [1:0]=count=1, [63:32]=lds_addr, [120:64]=global_addr, [127:126]=2
//   group1: [17:16]=data_size(2=4B), [20]=pad_en, [24:22]=pad_interval(3=16dw),
//           [31:25]=pad_amount(3=4dw), dim0=tile0=4096, dim1=tile1=1,
//           stride0=4096, workgroup_mask=0 (not in cluster)
// ---------------------------------------------------------------------------
__device__ __forceinline__ void tdm_load_w_tile(const _Float16* gsrc,
                                                unsigned lds_off) {
  const unsigned long long ga = (unsigned long long)(uintptr_t)gsrc;
  u32x4 g0;
  g0.x = 1u;                                                // count=1 (valid)
  g0.y = lds_off;                                           // lds_addr (bytes)
  g0.z = (unsigned)(ga & 0xFFFFFFFFu);                      // global_addr lo
  g0.w = (unsigned)((ga >> 32) & 0x01FFFFFFu) | (2u << 30); // addr[56:32]|type=2
  u32x8 g1;
  g1[0] = (2u << 16) | (1u << 20) | (3u << 22) | (3u << 25); // 4B, pad 4dw/16dw
  g1[1] = (4096u & 0xFFFFu) << 16;          // tensor_dim0 lo16
  g1[2] = (4096u >> 16) | (1u << 16);       // tensor_dim0 hi16 | tensor_dim1 lo16
  g1[3] = (4096u & 0xFFFFu) << 16;          // tensor_dim1 hi16 | tile_dim0=4096
  g1[4] = 1u;                               // tile_dim1=1, tile_dim2=0
  g1[5] = 4096u;                            // tensor_dim0_stride lo32
  g1[6] = 0u;                               // stride0 hi16 | stride1 lo16
  g1[7] = 0u;                               // stride1 hi32
  u32x4 gz = {0u, 0u, 0u, 0u};              // groups 2/3: dims beyond 2D unused
  asm volatile("tensor_load_to_lds %0, %1, %2, %3"
               :: "s"(g0), "s"(g1), "s"(gz), "s"(gz)
               : "memory");
}

// ---------------------------------------------------------------------------
// Implicit-GEMM TAda conv: D[co, pix] += W[co, ci|tap] * (alpha*x)[ci|tap, pix]
// Block: 256 threads = 8 wave32. M-tile 256 (all co), N-tile 64 pixels.
// Waves: 4 (M groups of 64) x 2 (N groups of 32); 8 WMMA tiles per wave.
// Weight tiles DMA'd by the Tensor Data Mover, double buffered.
// ---------------------------------------------------------------------------
__global__ __launch_bounds__(256) void tada_conv2d_wmma(
    const float* __restrict__ x,        // [B, CIN, T, H, W]
    const float* __restrict__ alpha,    // [B, CIN, T]
    const _Float16* __restrict__ wp,    // repacked weights
    float* __restrict__ out)            // [B, COUT, T, H, W]
{
  __shared__ __align__(64) _Float16 ldsW[2][COUT_][LDSPAD];   // 2 x 20.0 KB
  __shared__ __align__(64) _Float16 ldsX[2][PIXTILE][LDSPAD]; // 2 x  5.0 KB

  const int tid  = threadIdx.x;
  const int lane = tid & 31;
  const int wave = tid >> 5;
  const int mg   = wave >> 1;        // 0..3  -> co rows [mg*64, mg*64+64)
  const int ng   = wave & 1;         // 0..1  -> pixels  [ng*32, ng*32+32)
  const int half = lane >> 4;        // 0/1
  const int l15  = lane & 15;

  const int tile = blockIdx.x;       // 0..12
  const int bt   = blockIdx.y;       // 0..127
  const int b    = bt / T_;
  const int t    = bt % T_;
  const int pixBase = tile * PIXTILE;

  union Acc { v8f v; float f[8]; };
  Acc acc[4][2];
#pragma unroll
  for (int mt = 0; mt < 4; ++mt)
#pragma unroll
    for (int nt = 0; nt < 2; ++nt)
      acc[mt][nt].v = (v8f){0.f, 0.f, 0.f, 0.f, 0.f, 0.f, 0.f, 0.f};

  // X-staging role: thread -> (kk, 8 contiguous pixels)
  const int skk   = tid >> 3;              // 0..31
  const int jBase = (tid & 7) * 8;         // 0,8,...,56
  const size_t xbase = ((size_t)b * CIN_ * T_ + t) * HW_;

  // Stage the alpha-modulated, tap-shifted input tile for step s into buf.
  auto stageX = [&](int s, int buf) {
    const int k9 = s >> 3;
    const int cb = s & 7;
    const int dh = k9 / 3 - 1;
    const int dw = k9 % 3 - 1;
    const int ci = cb * 32 + skk;
    const float av = alpha[((size_t)b * CIN_ + ci) * T_ + t];
    const float* xp = x + xbase + (size_t)ci * T_ * HW_;
#pragma unroll
    for (int jj = 0; jj < 8; ++jj) {
      const int j = jBase + jj;
      const int p = pixBase + j;
      float v = 0.0f;
      if (p < HW_) {
        const int ph = p / W_ + dh;
        const int pw = p % W_ + dw;
        if (ph >= 0 && ph < H_ && pw >= 0 && pw < W_)
          v = xp[ph * W_ + pw] * av;
      }
      ldsX[buf][j][skk] = (_Float16)v;
    }
  };

  // ---- prologue: fill buffer 0 ----
  if (wave == 0)
    tdm_load_w_tile(wp, (unsigned)(uintptr_t)&ldsW[0][0][0]);
  stageX(0, 0);
  if (wave == 0)
    __builtin_amdgcn_s_wait_tensorcnt((short)0);
  __syncthreads();

  // ---- main pipelined loop over 72 k-steps ----
  for (int s = 0; s < NSTEP; ++s) {
    const int cur = s & 1;
    const int nxt = cur ^ 1;

    // issue next tile's DMA + stage next X while computing current step
    if (s + 1 < NSTEP) {
      if (wave == 0)
        tdm_load_w_tile(wp + (size_t)(s + 1) * COUT_ * CBLK,
                        (unsigned)(uintptr_t)&ldsW[nxt][0][0]);
      stageX(s + 1, nxt);
      if (s + 2 < NSTEP) {  // warm L2 two tiles ahead for the TDM
        const char* pre =
            (const char*)(wp + (size_t)(s + 2) * COUT_ * CBLK) + tid * 64;
        __builtin_prefetch(pre, 0, 1);
      }
    }

    // ---- fragments + 8 WMMAs per wave (read buffer `cur`) ----
    union Frag { v16h v; v8h h[2]; };
    Frag bf[2];
#pragma unroll
    for (int nt = 0; nt < 2; ++nt) {
      const int col = ng * 32 + nt * 16 + l15;     // pixel (N) index
      // B per lane: N = l15, K runs [8h,8h+8) and [8h+16,8h+24)
      bf[nt].h[0] = *(const v8h*)&ldsX[cur][col][half * 8];
      bf[nt].h[1] = *(const v8h*)&ldsX[cur][col][half * 8 + 16];
    }
#pragma unroll
    for (int mt = 0; mt < 4; ++mt) {
      const int row = mg * 64 + mt * 16 + l15;     // co (M) index
      // A per lane: M = l15, K runs [8h,8h+8) and [8h+16,8h+24)
      Frag af;
      af.h[0] = *(const v8h*)&ldsW[cur][row][half * 8];
      af.h[1] = *(const v8h*)&ldsW[cur][row][half * 8 + 16];
#pragma unroll
      for (int nt = 0; nt < 2; ++nt) {
        acc[mt][nt].v = __builtin_amdgcn_wmma_f32_16x16x32_f16(
            /*neg_a=*/false, af.v, /*neg_b=*/false, bf[nt].v,
            /*c_mod=*/(short)0, acc[mt][nt].v,
            /*reuse_a=*/false, /*reuse_b=*/false);
      }
    }

    // ensure next weight tile has landed (TDM in-order, wave0 tracks it),
    // then make next X tile + TDM data visible to all waves
    if (wave == 0)
      __builtin_amdgcn_s_wait_tensorcnt((short)0);
    __syncthreads();
  }

  // ---- store: C/D layout M = vgpr + 8*half, N = l15 ----
  const size_t outBase = ((size_t)b * COUT_ * T_ + t) * HW_;
#pragma unroll
  for (int nt = 0; nt < 2; ++nt) {
    const int pix = pixBase + ng * 32 + nt * 16 + l15;
    if (pix < HW_) {
#pragma unroll
      for (int mt = 0; mt < 4; ++mt) {
        const int coB = mg * 64 + mt * 16 + half * 8;
#pragma unroll
        for (int r = 0; r < 8; ++r) {
          out[outBase + (size_t)(coB + r) * T_ * HW_ + pix] = acc[mt][nt].f[r];
        }
      }
    }
  }
}

// ---------------------------------------------------------------------------
extern "C" void kernel_launch(void* const* d_in, const int* in_sizes, int n_in,
                              void* d_out, int out_size, void* d_ws, size_t ws_size,
                              hipStream_t stream) {
  (void)in_sizes; (void)n_in; (void)out_size; (void)ws_size;
  const float* x      = (const float*)d_in[0];  // [8,256,16,28,28]
  const float* alpha  = (const float*)d_in[1];  // [8,256,16,1,1]
  const float* weight = (const float*)d_in[2];  // [1,1,256,256,3,3]
  _Float16* wp = (_Float16*)d_ws;               // needs 9*8*256*32*2 = 1.18 MB

  // 9*8*256*32 = 589824 elements -> 2304 blocks of 256
  repack_weights_f16<<<2304, 256, 0, stream>>>(weight, wp);

  dim3 grid(NTILES, B_ * T_);
  tada_conv2d_wmma<<<grid, 256, 0, stream>>>(x, alpha, wp, (float*)d_out);
}
